// FeedforwardSNN_PulseGenerator_84980222919083
// MI455X (gfx1250) — matrix operand order repaired and checked by reading, not compiled, not run on hardware
//
#include <hip/hip_runtime.h>

typedef __attribute__((ext_vector_type(16))) _Float16     v16h;
typedef __attribute__((ext_vector_type(8)))  float        v8f;
typedef __attribute__((ext_vector_type(4)))  unsigned int v4u;
typedef __attribute__((ext_vector_type(8)))  unsigned int v8u;

#define NUM_STEPS 50
#define BETA      0.9f
#define THRESH    1.0f
#define WAVES     4                       // waves per workgroup (wave32)

// ---- LDS layout (bytes) ----
// [0, 32KB)        : 32 weight A-fragments (W2: f=0..15, W3: f=16..31), 1KB each
//                    per-lane 16B group0 at f*1024 + lane*16, group1 at +512
// [32KB, +2KB)     : 4 half-fragments for layer-1 [W1 | b1] (group1 is all-zero, not stored)
// [34KB, +2KB)     : 2 full fragments for Wo (row m=0 only)
// [36KB, ...)      : per-wave spike buffers: s1/s3 aliased (16 x 144B), s2 (16 x 272B)
#define WFRAG_BYTES    1024
#define W1B_BASE       32768
#define WO_BASE        (32768 + 4 * 512)
#define SPIKE_BASE     (WO_BASE + 2 * 1024)              // 36864
#define SROW13         144                                // 64 halfs + pad
#define SROW2          272                                // 128 halfs + pad
#define SPIKE_PER_WAVE (16 * SROW13 + 16 * SROW2)        // 6656
#define LDS_BYTES      (SPIKE_BASE + WAVES * SPIKE_PER_WAVE)  // 63488 < 64KB

__device__ __forceinline__ unsigned int pk2h(float a, float b) {
  return __builtin_bit_cast(unsigned int, __builtin_amdgcn_cvt_pkrtz(a, b));
}
__device__ __forceinline__ v16h frag_cast(v8u w) { return __builtin_bit_cast(v16h, w); }

__device__ __forceinline__ v8f wmma_f16(v16h a, v16h b, v8f c) {
  // D = A(16x32 f16) x B(32x16 f16) + C(16x16 f32)
  return __builtin_amdgcn_wmma_f32_16x16x32_f16(false, a, false, b, (short)0, c, false, false);
}

__device__ __forceinline__ v16h bias_frag(unsigned int p) {
  v8u w = {};
  w[0] = p;                 // A[m, K=0] = bias (lanes h==0 only); other K zero
  return frag_cast(w);
}

__device__ __forceinline__ v16h load_afrag(const unsigned char* lds, int f, int lane) {
  const unsigned char* p = lds + f * WFRAG_BYTES + lane * 16;
  v4u a = *reinterpret_cast<const v4u*>(p);
  v4u b = *reinterpret_cast<const v4u*>(p + 512);
  v8u w;
  w[0] = a[0]; w[1] = a[1]; w[2] = a[2]; w[3] = a[3];
  w[4] = b[0]; w[5] = b[1]; w[6] = b[2]; w[7] = b[3];
  return frag_cast(w);
}

__device__ __forceinline__ v16h load_w1b(const unsigned char* lds, int t, int lane) {
  // group1 (K=16..31) is structurally zero for the layer-1 fragment
  v4u a = *reinterpret_cast<const v4u*>(lds + W1B_BASE + t * 512 + lane * 16);
  v8u w = {};
  w[0] = a[0]; w[1] = a[1]; w[2] = a[2]; w[3] = a[3];
  return frag_cast(w);
}

__device__ __forceinline__ v16h load_wo(const unsigned char* lds, int c, int lane) {
  const unsigned char* p = lds + WO_BASE + c * WFRAG_BYTES + lane * 16;
  v4u a = *reinterpret_cast<const v4u*>(p);
  v4u b = *reinterpret_cast<const v4u*>(p + 512);
  v8u w;
  w[0] = a[0]; w[1] = a[1]; w[2] = a[2]; w[3] = a[3];
  w[4] = b[0]; w[5] = b[1]; w[6] = b[2]; w[7] = b[3];
  return frag_cast(w);
}

__device__ __forceinline__ v16h load_bfrag(const unsigned char* sb, int stride, int c,
                                           int l16, int h) {
  // B 32x16 f16: VGPRs 0-3 hold K=8h..8h+7, VGPRs 4-7 hold K=16+8h..16+8h+7 (2 halfs / VGPR)
  const unsigned char* p = sb + l16 * stride + 64 * c + 16 * h;
  v4u a = *reinterpret_cast<const v4u*>(p);
  v4u b = *reinterpret_cast<const v4u*>(p + 32);
  v8u w;
  w[0] = a[0]; w[1] = a[1]; w[2] = a[2]; w[3] = a[3];
  w[4] = b[0]; w[5] = b[1]; w[6] = b[2]; w[7] = b[3];
  return frag_cast(w);
}

// Computes spike once per element, stores packed f16 spikes to LDS, and returns the
// next-step carry  c = beta*m - spike*THRESH  (so no separate leak/reset pass is needed).
__device__ __forceinline__ v8f spike_carry(unsigned char* sb, int stride, int t,
                                           int l16, int h, const v8f& m) {
  float sp[8];
#pragma unroll
  for (int r = 0; r < 8; ++r) sp[r] = (m[r] > THRESH) ? THRESH : 0.0f;
  v4u u;
#pragma unroll
  for (int q = 0; q < 4; ++q) u[q] = pk2h(sp[2 * q], sp[2 * q + 1]);
  *reinterpret_cast<v4u*>(sb + l16 * stride + 32 * t + 16 * h) = u;
  v8f c;
#pragma unroll
  for (int r = 0; r < 8; ++r) c[r] = BETA * m[r] - sp[r];
  return c;
}

__device__ __forceinline__ float fast_tanh(float x) {
  float e = __expf(-2.0f * fabsf(x));
  float t = (1.0f - e) / (1.0f + e);
  return copysignf(t, x);
}

__global__ __launch_bounds__(WAVES * 32) void snn_pulse_kernel(
    const float* __restrict__ ts,
    const float* __restrict__ W1, const float* __restrict__ b1,
    const float* __restrict__ W2, const float* __restrict__ b2,
    const float* __restrict__ W3, const float* __restrict__ b3,
    const float* __restrict__ Wo, const float* __restrict__ bo,
    float* __restrict__ out, int Btot) {
  __shared__ __align__(16) unsigned char lds[LDS_BYTES];

  const int tid  = threadIdx.x;
  const int wave = tid >> 5;
  const int lane = tid & 31;
  const int l16  = lane & 15;
  const int h    = lane >> 4;

  // ---------------- prepack W2/W3 A-fragments (f32 -> f16) ----------------
  for (int f = wave; f < 32; f += WAVES) {
    const float* W;
    int t, c, ldw;
    if (f < 16) { W = W2; t = f >> 1;        c = f & 1;        ldw = 64;  }
    else        { W = W3; t = (f - 16) >> 2; c = (f - 16) & 3; ldw = 128; }
    const float* row = W + (t * 16 + l16) * ldw;      // A row m = output neuron
    int k0 = c * 32 + 8 * h;                           // VGPRs 0-3: K = k0..k0+7
    int k1 = k0 + 16;                                  // VGPRs 4-7: K = k1..k1+7
    v4u g0, g1;
#pragma unroll
    for (int q = 0; q < 4; ++q) {
      g0[q] = pk2h(row[k0 + 2 * q], row[k0 + 2 * q + 1]);
      g1[q] = pk2h(row[k1 + 2 * q], row[k1 + 2 * q + 1]);
    }
    *reinterpret_cast<v4u*>(lds + f * WFRAG_BYTES + lane * 16)       = g0;
    *reinterpret_cast<v4u*>(lds + f * WFRAG_BYTES + 512 + lane * 16) = g1;
  }
  // layer-1 [W1|b1] half-fragments (K=0:W1[:,0], K=1:W1[:,1], K=2:b1) and Wo fragments
  for (int f = 32 + wave; f < 38; f += WAVES) {
    if (f < 36) {
      int t = f - 32;
      int n = t * 16 + l16;
      v4u g0 = {};
      if (h == 0) {
        g0[0] = pk2h(W1[2 * n], W1[2 * n + 1]);
        g0[1] = pk2h(b1[n], 0.0f);
      }
      *reinterpret_cast<v4u*>(lds + W1B_BASE + t * 512 + lane * 16) = g0;
    } else {
      int c = f - 36;
      v4u g0 = {}, g1 = {};
      if (l16 == 0) {                                   // only row m = 0 is real
        int k0 = c * 32 + 8 * h, k1 = k0 + 16;
#pragma unroll
        for (int q = 0; q < 4; ++q) {
          g0[q] = pk2h(Wo[k0 + 2 * q], Wo[k0 + 2 * q + 1]);
          g1[q] = pk2h(Wo[k1 + 2 * q], Wo[k1 + 2 * q + 1]);
        }
      }
      *reinterpret_cast<v4u*>(lds + WO_BASE + c * WFRAG_BYTES + lane * 16)       = g0;
      *reinterpret_cast<v4u*>(lds + WO_BASE + c * WFRAG_BYTES + 512 + lane * 16) = g1;
    }
  }
  __syncthreads();

  const int b0 = (blockIdx.x * WAVES + wave) * 16;
  if (b0 >= Btot) return;                 // whole-wave uniform; no barriers below

  unsigned char* s13 = lds + SPIKE_BASE + wave * SPIKE_PER_WAVE;  // s1 & s3 (aliased)
  unsigned char* s2s = s13 + 16 * SROW13;                          // s2

  // bias fragments for layers 2/3/out (only K=0 of lanes h==0 nonzero)
  unsigned int ab2[8], ab3[4], abO;
#pragma unroll
  for (int t = 0; t < 8; ++t) ab2[t] = (h == 0) ? pk2h(b2[16 * t + l16], 0.0f) : 0u;
#pragma unroll
  for (int t = 0; t < 4; ++t) ab3[t] = (h == 0) ? pk2h(b3[16 * t + l16], 0.0f) : 0u;
  abO = (lane == 0) ? pk2h(bo[0], 0.0f) : 0u;

  // "ones" B fragment: B[K=0, n] = 1.0 for all batch columns (bias injection)
  v8u bu = {};
  bu[0] = (h == 0) ? 0x00003C00u : 0u;
  const v16h Bones = frag_cast(bu);

  // constant input B fragment for layer 1: K=0 -> x0, K=1 -> x1, K=2 -> 1.0 (bias)
  const float x0 = ts[(b0 + l16) * 2 + 0];
  const float x1 = ts[(b0 + l16) * 2 + 1];
  v8u bi = {};
  if (h == 0) {
    bi[0] = pk2h(x0, x1);
    bi[1] = 0x00003C00u;
  }
  const v16h Bin = frag_cast(bi);

  // persistent per-layer carries: c = beta*m - spike (m_next = c_prev + current)
  v8f c1[4] = {}, c2[8] = {}, c3[4] = {}, co = {};

  // ---------------- 50-step LIF recurrence, all state resident ----------------
#pragma unroll 1
  for (int step = 0; step < NUM_STEPS; ++step) {
    // layer 1: m1 = W1@x + b1 + carry   (input is constant rate-encoded state)
#pragma unroll
    for (int t = 0; t < 4; ++t) {
      v8f acc = wmma_f16(load_w1b(lds, t, lane), Bin, c1[t]);
      c1[t] = spike_carry(s13, SROW13, t, l16, h, acc);
    }

    // layer 2: m2 = W2 @ s1 + b2 + carry
    v16h B1[2];
    B1[0] = load_bfrag(s13, SROW13, 0, l16, h);
    B1[1] = load_bfrag(s13, SROW13, 1, l16, h);
#pragma unroll
    for (int t = 0; t < 8; ++t) {
      v8f acc = wmma_f16(bias_frag(ab2[t]), Bones, c2[t]);
      acc = wmma_f16(load_afrag(lds, 2 * t + 0, lane), B1[0], acc);
      acc = wmma_f16(load_afrag(lds, 2 * t + 1, lane), B1[1], acc);
      c2[t] = spike_carry(s2s, SROW2, t, l16, h, acc);
    }

    // layer 3: m3 = W3 @ s2 + b3 + carry
#pragma unroll
    for (int t = 0; t < 4; ++t) {
      v8f acc = wmma_f16(bias_frag(ab3[t]), Bones, c3[t]);
#pragma unroll
      for (int c = 0; c < 4; ++c)
        acc = wmma_f16(load_afrag(lds, 16 + 4 * t + c, lane),
                       load_bfrag(s2s, SROW2, c, l16, h), acc);
      c3[t] = spike_carry(s13, SROW13, t, l16, h, acc);   // s3 reuses s1 buffer
    }

    // output layer: mo = Wo @ s3 + bo + carry
    {
      v8f acc = wmma_f16(bias_frag(abO), Bones, co);
      acc = wmma_f16(load_wo(lds, 0, lane), load_bfrag(s13, SROW13, 0, l16, h), acc);
      acc = wmma_f16(load_wo(lds, 1, lane), load_bfrag(s13, SROW13, 1, l16, h), acc);
      // pulse from row m=0 (VGPR0 of lanes 0-15, b = lane)
      if (h == 0) out[(b0 + l16) * NUM_STEPS + step] = 5.0f * fast_tanh(acc[0]);
      // carry for output membrane (rows m>=1 stay exactly zero)
      v8f cn;
#pragma unroll
      for (int r = 0; r < 8; ++r) {
        float sp = (acc[r] > THRESH) ? THRESH : 0.0f;
        cn[r] = BETA * acc[r] - sp;
      }
      co = cn;
    }
  }
}

extern "C" void kernel_launch(void* const* d_in, const int* in_sizes, int n_in,
                              void* d_out, int out_size, void* d_ws, size_t ws_size,
                              hipStream_t stream) {
  (void)n_in; (void)out_size; (void)d_ws; (void)ws_size;
  const float* ts = (const float*)d_in[0];
  const float* W1 = (const float*)d_in[1];
  const float* b1 = (const float*)d_in[2];
  const float* W2 = (const float*)d_in[3];
  const float* b2 = (const float*)d_in[4];
  const float* W3 = (const float*)d_in[5];
  const float* b3 = (const float*)d_in[6];
  const float* Wo = (const float*)d_in[7];
  const float* bo = (const float*)d_in[8];
  float* out = (float*)d_out;

  int Btot = in_sizes[0] / 2;                    // target_state is (B, 2)
  int rowsPerBlock = WAVES * 16;                 // 16 batch rows per wave
  int blocks = (Btot + rowsPerBlock - 1) / rowsPerBlock;
  snn_pulse_kernel<<<dim3(blocks), dim3(WAVES * 32), 0, stream>>>(
      ts, W1, b1, W2, b2, W3, b3, Wo, bo, out, Btot);
}
